// FlowProposal_28020366639146
// MI455X (gfx1250) — matrix operand order repaired and checked by reading, not compiled, not run on hardware
//
#include <hip/hip_runtime.h>

// ---------------------------------------------------------------------------
// FlowProposal.log_prob on MI455X (gfx1250). Compute ~4.1 TFLOP/call, weights
// (~3 MB bf16) L2-resident. v_wmma_f32_16x16x32_bf16 everywhere; each B
// fragment register-reused across 2 row-tiles x (primal,tangent) = 4 WMMAs
// (64 FLOP/B vs L2) to stay under the L2 bandwidth roof. One 256-thread
// workgroup (8 wave32s) integrates 32 batch rows through all 40 RK4 stages
// with activations entirely in LDS/registers.
// ---------------------------------------------------------------------------

#define DIMN   256
#define HIDN   1024
#define BATCHN 16384
#define ROWS   32
#define NSTEPS 10
#define DT     (-0.1f)

typedef __attribute__((ext_vector_type(16))) __bf16 v16bf;
typedef __attribute__((ext_vector_type(8)))  float  v8f;

__device__ __forceinline__ unsigned int f2bf_u(float x) {
  union { float f; unsigned int u; } c; c.f = x;
  return (c.u + 0x7FFFu + ((c.u >> 16) & 1u)) >> 16;   // RNE fp32 -> bf16
}
__device__ __forceinline__ unsigned short f2bf(float x) { return (unsigned short)f2bf_u(x); }
__device__ __forceinline__ float bf2f(unsigned int h) {
  union { unsigned int u; float f; } c; c.u = h << 16; return c.f;
}

__device__ __forceinline__ float fast_tanh(float x) {
#if __has_builtin(__builtin_amdgcn_tanhf)
  return __builtin_amdgcn_tanhf(x);     // gfx1250 v_tanh_f32
#else
  return tanhf(x);
#endif
}

// A-fragment gather from row-major bf16 LDS tile, 32-wide K chunk kc.
// ISA 16-bit A layout: lanes 0-15 -> K 0..7/16..23; lanes 16-31 -> K 8..15/24..31.
__device__ __forceinline__ v16bf gatherA(const unsigned short* buf, int pitchDW,
                                         int m, int g, int kc) {
  const unsigned int* row = (const unsigned int*)buf + m * pitchDW;
  union { v16bf v; unsigned int u[8]; } f;
  int base = kc * 16 + 4 * g;
#pragma unroll
  for (int v = 0; v < 4; ++v) f.u[v] = row[base + v];
#pragma unroll
  for (int v = 4; v < 8; ++v) f.u[v] = row[base + 4 + v];
  return f.v;
}

// B fragment: pre-packed lane-major (1024B per 32x16 tile), 2x global_load_b128/lane.
__device__ __forceinline__ v16bf loadB(const unsigned int* wp, int kc, int ntg,
                                       int NtTiles, int lane) {
  const uint4* p = (const uint4*)(wp + ((((size_t)kc * NtTiles + ntg) << 8) + lane * 8));
  union { v16bf v; uint4 q[2]; } f;
  f.q[0] = p[0];
  f.q[1] = p[1];
  return f.v;
}

__device__ __forceinline__ v8f wmma_bf16(v16bf a, v16bf b, v8f c) {
  return __builtin_amdgcn_wmma_f32_16x16x32_bf16(false, a, false, b, (short)0, c,
                                                 false, false);
}

// ---------------------------------------------------------------------------
// Pack fp32 row-major (Ktot x Ntot) weight into WMMA B-fragment layout:
// tile(kt,nt) -> 256 dwords; lane = n%16 + 16*g; vgpr v -> K = 32kt + 8g + 16(v>>2) + 2(v&3).
// ---------------------------------------------------------------------------
__global__ void pack_bf16_wmma(const float* __restrict__ src,
                               unsigned int* __restrict__ dst,
                               int Ktot, int Ntot) {
  int Ntiles = Ntot >> 4;
  int total  = (Ktot >> 5) * Ntiles * 256;
  for (int id = blockIdx.x * blockDim.x + threadIdx.x; id < total;
       id += gridDim.x * blockDim.x) {
    int tile = id >> 8, within = id & 255;
    int lane = within >> 3, v = within & 7;
    int kt = tile / Ntiles, nt = tile % Ntiles;
    int n = (nt << 4) + (lane & 15);
    int g = lane >> 4;
    int k = (kt << 5) + (g << 3) + ((v & 4) << 2) + ((v & 3) << 1);
    unsigned int lo = f2bf_u(src[(size_t)k * Ntot + n]);
    unsigned int hi = f2bf_u(src[(size_t)(k + 1) * Ntot + n]);
    dst[id] = lo | (hi << 16);
  }
}

// ---------------------------------------------------------------------------
// Fused kernel: noise, 10 RK4 steps x 4 stages of (primal + JVP) MLP, output.
// 32 rows per workgroup = 2 row-tiles per wave; B fragments reused x4.
// ---------------------------------------------------------------------------
__global__ __launch_bounds__(256) void flow_logprob_kernel(
    const float* __restrict__ x,
    const float* __restrict__ W1,   // (257,1024): row 256 = t column, folded into bias
    const float* __restrict__ b1,
    const float* __restrict__ b2,
    const float* __restrict__ b3,
    const unsigned int* __restrict__ W1p,
    const unsigned int* __restrict__ W2p,
    const unsigned int* __restrict__ W3p,
    const unsigned int* __restrict__ Pp,
    float* __restrict__ out) {
  // padded pitches rotate LDS banks for strided row walks
  __shared__ __align__(16) unsigned short hb  [ROWS * 1032];  // L1 out (aliased by jvw)
  __shared__ __align__(16) unsigned short thb [ROWS * 1032];
  __shared__ __align__(16) unsigned short hb2 [ROWS * 1032];  // L2 out
  __shared__ __align__(16) unsigned short thb2[ROWS * 1032];
  __shared__ __align__(16) unsigned short xcb [ROWS * 264];
  __shared__ __align__(16) unsigned short nzb [ROWS * 264];
  __shared__ float part[8][33];
  __shared__ float trsum[ROWS];
  __shared__ float ljt[ROWS];
  float* jvw = (float*)hb;   // [32][260] f32, alias over dead L1 buffer

  const int tid  = threadIdx.x;
  const int lane = tid & 31;
  const int wave = tid >> 5;
  const int g    = lane >> 4;
  const int ln   = lane & 15;
  const int r0   = blockIdx.x * ROWS;

  // hoisted per-lane biases
  float b1v[8], wlv[8], b2v[8], b3v[2];
#pragma unroll
  for (int nt = 0; nt < 8; ++nt) {
    int n = wave * 128 + nt * 16 + ln;
    b1v[nt] = b1[n];
    wlv[nt] = W1[(size_t)256 * HIDN + n];   // t row of W1
    b2v[nt] = b2[n];
  }
#pragma unroll
  for (int nt = 0; nt < 2; ++nt) b3v[nt] = b3[wave * 32 + nt * 16 + ln];

  // per-lane-owned ODE state: element (m = rt*16 + r + 8g, n = wave*32 + nt*16 + ln)
  float xreg[2][2][8];
#pragma unroll
  for (int rt = 0; rt < 2; ++rt)
#pragma unroll
    for (int nt = 0; nt < 2; ++nt) {
      int n = wave * 32 + nt * 16 + ln;
#pragma unroll
      for (int r = 0; r < 8; ++r) {
        int m = rt * 16 + r + 8 * g;
        float v = x[(size_t)(r0 + m) * DIMN + n];
        xreg[rt][nt][r] = v;
        xcb[m * 264 + n] = f2bf(v);
      }
    }
  if (tid < ROWS) ljt[tid] = 0.0f;
  __syncthreads();

  // ---- noise = sin(1000 * (x @ proj)) * sqrt(2) ----
  {
    v8f acc[2][2];
#pragma unroll
    for (int rt = 0; rt < 2; ++rt)
#pragma unroll
      for (int nt = 0; nt < 2; ++nt)
#pragma unroll
        for (int r = 0; r < 8; ++r) acc[rt][nt][r] = 0.0f;
    for (int kc = 0; kc < DIMN / 32; ++kc) {
      v16bf a0 = gatherA(xcb, 132, ln, g, kc);
      v16bf a1 = gatherA(xcb, 132, 16 + ln, g, kc);
#pragma unroll
      for (int nt = 0; nt < 2; ++nt) {
        v16bf b = loadB(Pp, kc, wave * 2 + nt, 16, lane);
        acc[0][nt] = wmma_bf16(a0, b, acc[0][nt]);
        acc[1][nt] = wmma_bf16(a1, b, acc[1][nt]);
      }
    }
#pragma unroll
    for (int rt = 0; rt < 2; ++rt)
#pragma unroll
      for (int nt = 0; nt < 2; ++nt) {
        int n = wave * 32 + nt * 16 + ln;
#pragma unroll
        for (int r = 0; r < 8; ++r) {
          int m = rt * 16 + r + 8 * g;
          float nz = __sinf(acc[rt][nt][r] * 1000.0f) * 1.41421356237f;
          nzb[m * 264 + n] = f2bf(nz);
        }
      }
  }
  __syncthreads();

  float vsum[2][2][8];

  for (int step = 0; step < NSTEPS; ++step) {
    float t0 = 1.0f + DT * (float)step;
    if (tid < ROWS) trsum[tid] = 0.0f;
#pragma unroll
    for (int rt = 0; rt < 2; ++rt)
#pragma unroll
      for (int nt = 0; nt < 2; ++nt)
#pragma unroll
        for (int r = 0; r < 8; ++r) vsum[rt][nt][r] = 0.0f;

    for (int stage = 0; stage < 4; ++stage) {
      float tc  = (stage == 0) ? t0 : (stage == 3 ? t0 + DT : t0 + 0.5f * DT);
      float wgt = (stage == 1 || stage == 2) ? 2.0f : 1.0f;

      // ---- Layer 1: [xc,t] -> 1024 (primal + tangent), two halves of 4 N-tiles ----
#pragma unroll
      for (int half = 0; half < 2; ++half) {
        v8f accp[2][4], acct[2][4];
#pragma unroll
        for (int j = 0; j < 4; ++j) {
          float bias = b1v[half * 4 + j] + tc * wlv[half * 4 + j];
#pragma unroll
          for (int rt = 0; rt < 2; ++rt)
#pragma unroll
            for (int r = 0; r < 8; ++r) { accp[rt][j][r] = bias; acct[rt][j][r] = 0.0f; }
        }
        for (int kc = 0; kc < DIMN / 32; ++kc) {
          v16bf ap0 = gatherA(xcb, 132, ln, g, kc);
          v16bf ap1 = gatherA(xcb, 132, 16 + ln, g, kc);
          v16bf an0 = gatherA(nzb, 132, ln, g, kc);
          v16bf an1 = gatherA(nzb, 132, 16 + ln, g, kc);
#pragma unroll
          for (int j = 0; j < 4; ++j) {
            v16bf b = loadB(W1p, kc, wave * 8 + half * 4 + j, 64, lane);
            accp[0][j] = wmma_bf16(ap0, b, accp[0][j]);
            accp[1][j] = wmma_bf16(ap1, b, accp[1][j]);
            acct[0][j] = wmma_bf16(an0, b, acct[0][j]);
            acct[1][j] = wmma_bf16(an1, b, acct[1][j]);
          }
        }
#pragma unroll
        for (int rt = 0; rt < 2; ++rt)
#pragma unroll
          for (int j = 0; j < 4; ++j) {
            int n = wave * 128 + (half * 4 + j) * 16 + ln;
#pragma unroll
            for (int r = 0; r < 8; ++r) {
              int m = rt * 16 + r + 8 * g;
              float h  = fast_tanh(accp[rt][j][r]);
              float th = (1.0f - h * h) * acct[rt][j][r];
              hb [m * 1032 + n] = f2bf(h);
              thb[m * 1032 + n] = f2bf(th);
            }
          }
      }
      __syncthreads();   // h1/th1 complete

      // ---- Layer 2: 1024 -> 1024, double-buffered output (hb2/thb2) ----
#pragma unroll
      for (int half = 0; half < 2; ++half) {
        v8f accp[2][4], acct[2][4];
#pragma unroll
        for (int j = 0; j < 4; ++j) {
          float bias = b2v[half * 4 + j];
#pragma unroll
          for (int rt = 0; rt < 2; ++rt)
#pragma unroll
            for (int r = 0; r < 8; ++r) { accp[rt][j][r] = bias; acct[rt][j][r] = 0.0f; }
        }
        for (int kc = 0; kc < HIDN / 32; ++kc) {
          v16bf ap0 = gatherA(hb,  516, ln, g, kc);
          v16bf ap1 = gatherA(hb,  516, 16 + ln, g, kc);
          v16bf at0 = gatherA(thb, 516, ln, g, kc);
          v16bf at1 = gatherA(thb, 516, 16 + ln, g, kc);
#pragma unroll
          for (int j = 0; j < 4; ++j) {
            v16bf b = loadB(W2p, kc, wave * 8 + half * 4 + j, 64, lane);
            accp[0][j] = wmma_bf16(ap0, b, accp[0][j]);
            accp[1][j] = wmma_bf16(ap1, b, accp[1][j]);
            acct[0][j] = wmma_bf16(at0, b, acct[0][j]);
            acct[1][j] = wmma_bf16(at1, b, acct[1][j]);
          }
        }
        // hb2/thb2 not read by anyone during layer 2: store without a barrier
#pragma unroll
        for (int rt = 0; rt < 2; ++rt)
#pragma unroll
          for (int j = 0; j < 4; ++j) {
            int n = wave * 128 + (half * 4 + j) * 16 + ln;
#pragma unroll
            for (int r = 0; r < 8; ++r) {
              int m = rt * 16 + r + 8 * g;
              float h  = fast_tanh(accp[rt][j][r]);
              float th = (1.0f - h * h) * acct[rt][j][r];
              hb2 [m * 1032 + n] = f2bf(h);
              thb2[m * 1032 + n] = f2bf(th);
            }
          }
      }
      __syncthreads();   // h2/th2 complete; hb now dead -> jvw may reuse it

      // ---- Layer 3: 1024 -> 256, velocity v and JVP jv ----
      v8f accv[2][2], accj[2][2];
#pragma unroll
      for (int rt = 0; rt < 2; ++rt)
#pragma unroll
        for (int nt = 0; nt < 2; ++nt)
#pragma unroll
          for (int r = 0; r < 8; ++r) { accv[rt][nt][r] = b3v[nt]; accj[rt][nt][r] = 0.0f; }
      for (int kc = 0; kc < HIDN / 32; ++kc) {
        v16bf ap0 = gatherA(hb2,  516, ln, g, kc);
        v16bf ap1 = gatherA(hb2,  516, 16 + ln, g, kc);
        v16bf at0 = gatherA(thb2, 516, ln, g, kc);
        v16bf at1 = gatherA(thb2, 516, 16 + ln, g, kc);
#pragma unroll
        for (int nt = 0; nt < 2; ++nt) {
          v16bf b = loadB(W3p, kc, wave * 2 + nt, 16, lane);
          accv[0][nt] = wmma_bf16(ap0, b, accv[0][nt]);
          accv[1][nt] = wmma_bf16(ap1, b, accv[1][nt]);
          accj[0][nt] = wmma_bf16(at0, b, accj[0][nt]);
          accj[1][nt] = wmma_bf16(at1, b, accj[1][nt]);
        }
      }

      // RK4 epilogue: owner-exclusive updates (all waves already past hb/xcb reads)
      float acoef = (stage <= 1) ? (0.5f * DT) : DT;
#pragma unroll
      for (int rt = 0; rt < 2; ++rt)
#pragma unroll
        for (int nt = 0; nt < 2; ++nt) {
          int n = wave * 32 + nt * 16 + ln;
#pragma unroll
          for (int r = 0; r < 8; ++r) {
            int m = rt * 16 + r + 8 * g;
            float v = accv[rt][nt][r];
            vsum[rt][nt][r] += wgt * v;
            float nz = bf2f(nzb[m * 264 + n]);
            jvw[m * 260 + n] = nz * accj[rt][nt][r];
            if (stage < 3) {
              xcb[m * 264 + n] = f2bf(xreg[rt][nt][r] + acoef * v);
            } else {
              float xn = xreg[rt][nt][r] + (DT / 6.0f) * vsum[rt][nt][r];
              xreg[rt][nt][r] = xn;
              xcb[m * 264 + n] = f2bf(xn);
            }
          }
        }
      __syncthreads();

      // Hutchinson trace: two-phase row reduction of noise .* jv
      {
        int m = tid & 31, chunk = tid >> 5;
        float s = 0.0f;
        const float* rowp = jvw + m * 260 + chunk * 32;
#pragma unroll
        for (int c = 0; c < 32; ++c) s += rowp[c];
        part[chunk][m] = s;
      }
      __syncthreads();
      if (tid < ROWS) {
        float s = 0.0f;
#pragma unroll
        for (int c = 0; c < 8; ++c) s += part[c][tid];
        trsum[tid] += wgt * s;
      }
      __syncthreads();
    }  // stage
    if (tid < ROWS) ljt[tid] += (DT / 6.0f) * trsum[tid];
    __syncthreads();
  }  // step

  // ---- output: -0.5*||x0||^2 - 0.5*DIM*log(2pi) - ljt ----
#pragma unroll
  for (int rt = 0; rt < 2; ++rt)
#pragma unroll
    for (int nt = 0; nt < 2; ++nt) {
      int n = wave * 32 + nt * 16 + ln;
#pragma unroll
      for (int r = 0; r < 8; ++r) {
        int m = rt * 16 + r + 8 * g;
        float v = xreg[rt][nt][r];
        jvw[m * 260 + n] = v * v;
      }
    }
  __syncthreads();
  {
    int m = tid & 31, chunk = tid >> 5;
    float s = 0.0f;
    const float* rowp = jvw + m * 260 + chunk * 32;
#pragma unroll
    for (int c = 0; c < 32; ++c) s += rowp[c];
    part[chunk][m] = s;
  }
  __syncthreads();
  if (tid < ROWS) {
    float s = 0.0f;
#pragma unroll
    for (int c = 0; c < 8; ++c) s += part[c][tid];
    const float LOG2PI = 1.8378770664093453f;
    out[r0 + tid] = -0.5f * s - 0.5f * (float)DIMN * LOG2PI - ljt[tid];
  }
}

// ---------------------------------------------------------------------------
extern "C" void kernel_launch(void* const* d_in, const int* in_sizes, int n_in,
                              void* d_out, int out_size, void* d_ws, size_t ws_size,
                              hipStream_t stream) {
  const float* x  = (const float*)d_in[0];
  const float* W1 = (const float*)d_in[1];
  const float* b1 = (const float*)d_in[2];
  const float* W2 = (const float*)d_in[3];
  const float* b2 = (const float*)d_in[4];
  const float* W3 = (const float*)d_in[5];
  const float* b3 = (const float*)d_in[6];
  const float* pj = (const float*)d_in[7];
  float* out = (float*)d_out;

  // workspace layout (dwords): packed bf16 B-fragment weights (~3.2 MB total)
  unsigned int* ws  = (unsigned int*)d_ws;
  unsigned int* W1p = ws;               // (256/32)*(1024/16)*256  = 131072
  unsigned int* W2p = W1p + 131072;     // (1024/32)*(1024/16)*256 = 524288
  unsigned int* W3p = W2p + 524288;     // (1024/32)*(256/16)*256  = 131072
  unsigned int* Pp  = W3p + 131072;     // (256/32)*(256/16)*256   = 32768

  pack_bf16_wmma<<<256,  256, 0, stream>>>(W1, W1p, 256, 1024);  // rows 0..255 only
  pack_bf16_wmma<<<1024, 256, 0, stream>>>(W2, W2p, 1024, 1024);
  pack_bf16_wmma<<<256,  256, 0, stream>>>(W3, W3p, 1024, 256);
  pack_bf16_wmma<<<64,   256, 0, stream>>>(pj, Pp, 256, 256);

  flow_logprob_kernel<<<BATCHN / ROWS, 256, 0, stream>>>(
      x, W1, b1, b2, b3, W1p, W2p, W3p, Pp, out);
}
